// MammothModa2Qwen2ForCausalLM_84739704750514
// MI455X (gfx1250) — matrix operand order repaired and checked by reading, not compiled, not run on hardware
//
#include <hip/hip_runtime.h>
#include <hip/hip_bf16.h>

// ---------------------------------------------------------------------------
// Problem constants (from reference): B=4, L=4096, D=2048, I=5632
// ---------------------------------------------------------------------------
#define DD   2048
#define II   5632
#define NTOK 16384   // B*L

typedef __attribute__((ext_vector_type(16))) __bf16 v16bf;
typedef __attribute__((ext_vector_type(8)))  __bf16 v8bf;
typedef __attribute__((ext_vector_type(8)))  float  v8f;

static __device__ __forceinline__ __bf16 f2bf(float f) { return (__bf16)f; }

// Assemble a 16-element bf16 WMMA fragment from two contiguous 8-elem chunks.
static __device__ __forceinline__ v16bf frag2(const __bf16* p0, const __bf16* p1) {
  v8bf lo = *(const v8bf*)p0;
  v8bf hi = *(const v8bf*)p1;
  return __builtin_shufflevector(lo, hi, 0,1,2,3,4,5,6,7,8,9,10,11,12,13,14,15);
}

#define WMMA_BF16(A, B, C) \
  __builtin_amdgcn_wmma_f32_16x16x32_bf16(false, (A), false, (B), (short)0, (C), false, false)

// ---------------------------------------------------------------------------
// Stage 1a: fp32 -> bf16 elementwise (activations)
// ---------------------------------------------------------------------------
__global__ void k_cvt_x(const float* __restrict__ in, __bf16* __restrict__ out, int n) {
  int i = blockIdx.x * blockDim.x + threadIdx.x;
  if (i < n) out[i] = f2bf(in[i]);
}

// ---------------------------------------------------------------------------
// Stage 1b: fp32 (R,C) -> bf16 (C,R) transpose-convert (weights), so the
// WMMA B-fragment K dimension becomes the contiguous axis.
// ---------------------------------------------------------------------------
__global__ void k_transpose_bf(const float* __restrict__ in, __bf16* __restrict__ out,
                               int R, int C) {
  int i = blockIdx.x * blockDim.x + threadIdx.x;
  if (i < R * C) {
    int r = i / C, c = i % C;
    out[(size_t)c * R + r] = f2bf(in[i]);
  }
}

// ---------------------------------------------------------------------------
// Stage 2: gate/up projections for BOTH experts + fused silu*up + per-token
// expert select.  Grid: (II/64, NTOK/32), 128 threads = 4 waves.
// Each wave: 32x16 output (2 M-subtiles), 4 weight streams -> 8 accumulators,
// 8 WMMAs per 12 b128 loads per K-step.
//   x   : (NTOK, DD)  bf16
//   *T  : (II, DD)    bf16 (transposed weights, K=D contiguous)
//   h   : (NTOK, II)  bf16 (selected expert's silu(g)*u)
// ---------------------------------------------------------------------------
__global__ __launch_bounds__(128) void k_gateup(
    const __bf16* __restrict__ x,
    const __bf16* __restrict__ ugT, const __bf16* __restrict__ uuT,
    const __bf16* __restrict__ ggT, const __bf16* __restrict__ guT,
    const unsigned char* __restrict__ mask,
    __bf16* __restrict__ h)
{
  const int lane = threadIdx.x & 31;
  const int wave = threadIdx.x >> 5;
  const int half = lane >> 4;     // 0: lanes 0-15, 1: lanes 16-31
  const int ln   = lane & 15;

  const int m0  = blockIdx.y * 32;
  const int col = blockIdx.x * 64 + wave * 16 + ln;

  v8f c_ug0 = {}, c_uu0 = {}, c_gg0 = {}, c_gu0 = {};
  v8f c_ug1 = {}, c_uu1 = {}, c_gg1 = {}, c_gu1 = {};

  // A fragments (16x32 bf16): row = ln (+16), K chunks at k+8*half, k+16+8*half
  const __bf16* a0 = x + (size_t)(m0 + ln)      * DD + half * 8;
  const __bf16* a1 = x + (size_t)(m0 + 16 + ln) * DD + half * 8;
  // B fragments (32x16 bf16): col = ln, K chunks at k+8*half, k+16+8*half
  const __bf16* b0 = ugT + (size_t)col * DD + half * 8;
  const __bf16* b1 = uuT + (size_t)col * DD + half * 8;
  const __bf16* b2 = ggT + (size_t)col * DD + half * 8;
  const __bf16* b3 = guT + (size_t)col * DD + half * 8;

  for (int k = 0; k < DD; k += 32) {
    v16bf A0 = frag2(a0 + k, a0 + k + 16);
    v16bf A1 = frag2(a1 + k, a1 + k + 16);
    v16bf F0 = frag2(b0 + k, b0 + k + 16);
    v16bf F1 = frag2(b1 + k, b1 + k + 16);
    v16bf F2 = frag2(b2 + k, b2 + k + 16);
    v16bf F3 = frag2(b3 + k, b3 + k + 16);
    c_ug0 = WMMA_BF16(A0, F0, c_ug0);
    c_ug1 = WMMA_BF16(A1, F0, c_ug1);
    c_uu0 = WMMA_BF16(A0, F1, c_uu0);
    c_uu1 = WMMA_BF16(A1, F1, c_uu1);
    c_gg0 = WMMA_BF16(A0, F2, c_gg0);
    c_gg1 = WMMA_BF16(A1, F2, c_gg1);
    c_gu0 = WMMA_BF16(A0, F3, c_gu0);
    c_gu1 = WMMA_BF16(A1, F3, c_gu1);
  }

  // C layout: VGPR r holds (row = r + 8*half, col = ln). Fused epilogue:
  // pick expert per token row, silu(gate)*up, store selected h in bf16.
  #pragma unroll
  for (int r = 0; r < 8; ++r) {
    {
      int row = m0 + r + half * 8;
      float g, u;
      if (mask[row]) { g = c_gg0[r]; u = c_gu0[r]; }
      else           { g = c_ug0[r]; u = c_uu0[r]; }
      float s = g / (1.0f + __expf(-g));   // silu
      h[(size_t)row * II + col] = f2bf(s * u);
    }
    {
      int row = m0 + 16 + r + half * 8;
      float g, u;
      if (mask[row]) { g = c_gg1[r]; u = c_gu1[r]; }
      else           { g = c_ug1[r]; u = c_uu1[r]; }
      float s = g / (1.0f + __expf(-g));
      h[(size_t)row * II + col] = f2bf(s * u);
    }
  }
}

// ---------------------------------------------------------------------------
// Stage 3: down projection for both experts + per-token select -> fp32 out.
// Grid: (DD/64, NTOK/64), 128 threads = 4 waves.
// Each wave: 64x16 output (4 M-subtiles), 2 weight streams -> 8 accumulators,
// 8 WMMAs per 12 b128 loads per K-step.
//   h    : (NTOK, II) bf16
//   *dT  : (DD, II)   bf16 (transposed down weights, K=I contiguous)
//   out  : (NTOK, DD) fp32
// ---------------------------------------------------------------------------
__global__ __launch_bounds__(128) void k_down(
    const __bf16* __restrict__ h,
    const __bf16* __restrict__ udT, const __bf16* __restrict__ gdT,
    const unsigned char* __restrict__ mask,
    float* __restrict__ out)
{
  const int lane = threadIdx.x & 31;
  const int wave = threadIdx.x >> 5;
  const int half = lane >> 4;
  const int ln   = lane & 15;

  const int m0  = blockIdx.y * 64;
  const int col = blockIdx.x * 64 + wave * 16 + ln;

  v8f cu0 = {}, cu1 = {}, cu2 = {}, cu3 = {};
  v8f cg0 = {}, cg1 = {}, cg2 = {}, cg3 = {};

  const __bf16* a0 = h + (size_t)(m0 + ln)      * II + half * 8;
  const __bf16* a1 = h + (size_t)(m0 + 16 + ln) * II + half * 8;
  const __bf16* a2 = h + (size_t)(m0 + 32 + ln) * II + half * 8;
  const __bf16* a3 = h + (size_t)(m0 + 48 + ln) * II + half * 8;
  const __bf16* bu = udT + (size_t)col * II + half * 8;
  const __bf16* bg = gdT + (size_t)col * II + half * 8;

  for (int k = 0; k < II; k += 32) {
    v16bf A0 = frag2(a0 + k, a0 + k + 16);
    v16bf A1 = frag2(a1 + k, a1 + k + 16);
    v16bf A2 = frag2(a2 + k, a2 + k + 16);
    v16bf A3 = frag2(a3 + k, a3 + k + 16);
    v16bf FU = frag2(bu + k, bu + k + 16);
    v16bf FG = frag2(bg + k, bg + k + 16);
    cu0 = WMMA_BF16(A0, FU, cu0);
    cu1 = WMMA_BF16(A1, FU, cu1);
    cu2 = WMMA_BF16(A2, FU, cu2);
    cu3 = WMMA_BF16(A3, FU, cu3);
    cg0 = WMMA_BF16(A0, FG, cg0);
    cg1 = WMMA_BF16(A1, FG, cg1);
    cg2 = WMMA_BF16(A2, FG, cg2);
    cg3 = WMMA_BF16(A3, FG, cg3);
  }

  #pragma unroll
  for (int r = 0; r < 8; ++r) {
    int row0 = m0 + r + half * 8;
    out[(size_t)(row0)      * DD + col] = mask[row0]      ? cg0[r] : cu0[r];
    out[(size_t)(row0 + 16) * DD + col] = mask[row0 + 16] ? cg1[r] : cu1[r];
    out[(size_t)(row0 + 32) * DD + col] = mask[row0 + 32] ? cg2[r] : cu2[r];
    out[(size_t)(row0 + 48) * DD + col] = mask[row0 + 48] ? cg3[r] : cu3[r];
  }
}

// ---------------------------------------------------------------------------
// Host launcher
// ---------------------------------------------------------------------------
extern "C" void kernel_launch(void* const* d_in, const int* in_sizes, int n_in,
                              void* d_out, int out_size, void* d_ws, size_t ws_size,
                              hipStream_t stream) {
  (void)in_sizes; (void)n_in; (void)out_size; (void)ws_size;

  const float*         x    = (const float*)d_in[0];
  const unsigned char* mask = (const unsigned char*)d_in[1];  // jax bool -> byte
  const float* ug = (const float*)d_in[2];
  const float* uu = (const float*)d_in[3];
  const float* ud = (const float*)d_in[4];
  const float* gg = (const float*)d_in[5];
  const float* gu = (const float*)d_in[6];
  const float* gd = (const float*)d_in[7];

  // Workspace carve-up (all chunks 16B aligned; total ~390 MB)
  char* ws = (char*)d_ws;
  size_t off = 0;
  const size_t W = (size_t)DD * II;          // 11.5M elems per weight matrix
  __bf16* xbf = (__bf16*)(ws + off); off += (size_t)NTOK * DD * 2;
  __bf16* ugT = (__bf16*)(ws + off); off += W * 2;
  __bf16* uuT = (__bf16*)(ws + off); off += W * 2;
  __bf16* ggT = (__bf16*)(ws + off); off += W * 2;
  __bf16* guT = (__bf16*)(ws + off); off += W * 2;
  __bf16* udT = (__bf16*)(ws + off); off += W * 2;
  __bf16* gdT = (__bf16*)(ws + off); off += W * 2;
  __bf16* hbf = (__bf16*)(ws + off); off += (size_t)NTOK * II * 2;

  // Stage 1: convert / transpose-convert to bf16
  {
    int n = NTOK * DD;
    k_cvt_x<<<(n + 255) / 256, 256, 0, stream>>>(x, xbf, n);
    int nw = (int)W;
    int gsz = (nw + 255) / 256;
    k_transpose_bf<<<gsz, 256, 0, stream>>>(ug, ugT, DD, II);  // (D,I)->(I,D)
    k_transpose_bf<<<gsz, 256, 0, stream>>>(uu, uuT, DD, II);
    k_transpose_bf<<<gsz, 256, 0, stream>>>(gg, ggT, DD, II);
    k_transpose_bf<<<gsz, 256, 0, stream>>>(gu, guT, DD, II);
    k_transpose_bf<<<gsz, 256, 0, stream>>>(ud, udT, II, DD);  // (I,D)->(D,I)
    k_transpose_bf<<<gsz, 256, 0, stream>>>(gd, gdT, II, DD);
  }

  // Stage 2: gate/up WMMA GEMMs (both experts) + fused silu*up + select
  {
    dim3 grid(II / 64, NTOK / 32);   // 88 x 512
    k_gateup<<<grid, 128, 0, stream>>>(xbf, ugT, uuT, ggT, guT, mask, hbf);
  }

  // Stage 3: down WMMA GEMMs (both experts) + select -> fp32 output
  {
    dim3 grid(DD / 64, NTOK / 64);   // 32 x 256
    k_down<<<grid, 128, 0, stream>>>(hbf, udT, gdT, mask, (float*)d_out);
  }
}
